// SPMLP_13005160973281
// MI455X (gfx1250) — matrix operand order repaired and checked by reading, not compiled, not run on hardware
//
#include <hip/hip_runtime.h>
#include <hip/hip_bf16.h>
#include <math.h>

// Problem constants (fixed by the reference's setup_inputs).
#define TOK   4096      // B*S tokens
#define DDIM  1024      // hidden dim
#define NEXP  8         // experts
#define IDIM  4096      // intermediate dim
#define EPS2  0.02f     // 2 * jitter_eps

typedef __attribute__((ext_vector_type(16))) _Float16 v16h;
typedef __attribute__((ext_vector_type(8)))  _Float16 v8h;
typedef __attribute__((ext_vector_type(4)))  _Float16 v4h;
typedef __attribute__((ext_vector_type(8)))  float    v8f;
typedef __attribute__((ext_vector_type(4)))  float    v4f;

// ---------------------------------------------------------------------------
// Kernel 1: router GEMM + sparsemixer top-2 + per-expert gather-list build.
// One wave32 per token. 256 threads/block = 8 tokens/block.
// ---------------------------------------------------------------------------
__global__ void router_kernel(const float* __restrict__ x,
                              const float* __restrict__ gw,
                              float* __restrict__ logits_out,
                              int*   __restrict__ cnt,
                              int*   __restrict__ tokList,
                              float* __restrict__ wList) {
    const int wave = threadIdx.x >> 5;
    const int lane = threadIdx.x & 31;
    const int t = blockIdx.x * 8 + wave;
    if (t >= TOK) return;

    const float* xr = x + (size_t)t * DDIM;
    float acc[NEXP];
#pragma unroll
    for (int e = 0; e < NEXP; ++e) acc[e] = 0.f;

    for (int k = lane; k < DDIM; k += 32) {
        const float xv = xr[k];
#pragma unroll
        for (int e = 0; e < NEXP; ++e) acc[e] += xv * gw[e * DDIM + k];
    }
#pragma unroll
    for (int off = 16; off > 0; off >>= 1) {
#pragma unroll
        for (int e = 0; e < NEXP; ++e) acc[e] += __shfl_xor(acc[e], off, 32);
    }

    if (lane < NEXP) logits_out[t * NEXP + lane] = acc[lane];

    if (lane == 0) {
        // ---- top-1 ----
        float top1 = acc[0]; int sel1 = 0;
        for (int e = 1; e < NEXP; ++e) if (acc[e] > top1) { top1 = acc[e]; sel1 = e; }
        float sum1 = 0.f;
        for (int e = 0; e < NEXP; ++e) {
            const float factor = fmaxf(fabsf(acc[e]), top1);
            const bool masked = ((top1 - acc[e]) / factor) > EPS2;
            if (!masked) sum1 += __expf(acc[e] - top1);
        }
        const float m1 = 1.0f / sum1;

        // ---- top-2 (exclude sel1) ----
        float top2 = -INFINITY; int sel2 = 0;
        for (int e = 0; e < NEXP; ++e) {
            if (e == sel1) continue;
            if (acc[e] > top2) { top2 = acc[e]; sel2 = e; }
        }
        float sum2 = 0.f;
        for (int e = 0; e < NEXP; ++e) {
            if (e == sel1) continue;
            const float factor = fmaxf(fabsf(acc[e]), top2);
            const bool masked = ((top2 - acc[e]) / factor) > EPS2;
            if (!masked) sum2 += __expf(acc[e] - top2);
        }
        const float m2 = 1.0f / sum2;

        int s1 = atomicAdd(&cnt[sel1], 1);
        tokList[sel1 * TOK + s1] = t;  wList[sel1 * TOK + s1] = m1;
        int s2 = atomicAdd(&cnt[sel2], 1);
        tokList[sel2 * TOK + s2] = t;  wList[sel2 * TOK + s2] = m2;
    }
}

// ---------------------------------------------------------------------------
// Weight pre-conversion: f32 -> f16, one matrix per launch (memory bound).
// ---------------------------------------------------------------------------
__global__ __launch_bounds__(256)
void cvt_f32_to_f16(const float* __restrict__ src, _Float16* __restrict__ dst, int n4) {
    const int i = blockIdx.x * blockDim.x + threadIdx.x;
    if (i < n4) {
        const v4f f = ((const v4f*)src)[i];
        v4h h;
        h[0] = (_Float16)f[0]; h[1] = (_Float16)f[1];
        h[2] = (_Float16)f[2]; h[3] = (_Float16)f[3];
        ((v4h*)dst)[i] = h;
    }
}

// ---------------------------------------------------------------------------
// Fragment helpers (contiguous-K layouts, wave32)
// ---------------------------------------------------------------------------
__device__ inline v16h cvt16_from_f32(const float* __restrict__ p) {
    v16h r;
#pragma unroll
    for (int q = 0; q < 4; ++q) {
        const v4f f = *(const v4f*)(p + 4 * q);
        r[4 * q + 0] = (_Float16)f[0];
        r[4 * q + 1] = (_Float16)f[1];
        r[4 * q + 2] = (_Float16)f[2];
        r[4 * q + 3] = (_Float16)f[3];
    }
    return r;
}

// B fragment (32x16): lane n=l&15, K half = l>>4; 16 consecutive K elements.
__device__ inline v16h load_bfrag(const float* __restrict__ p)    { return cvt16_from_f32(p); }
__device__ inline v16h load_bfrag(const _Float16* __restrict__ p) { return *(const v16h*)p; }

// A fragment (16x32 f16) from an LDS row (row-major, K contiguous):
//   VGPR0..7 : K = k0 + hi*16 + {0..15}
__device__ inline v16h lds_afrag(const _Float16* __restrict__ row, int k0, int hi) {
    const v8h lo = *(const v8h*)(row + k0 + hi * 16);
    const v8h hh = *(const v8h*)(row + k0 + hi * 16 + 8);
    return __builtin_shufflevector(lo, hh, 0, 1, 2, 3, 4, 5, 6, 7,
                                           8, 9, 10, 11, 12, 13, 14, 15);
}

// ---------------------------------------------------------------------------
// Kernel 2: fused expert FFN.
//   grid = (TOK/32, NEXP), block = 512 (16 waves).
//   BM=32 gathered tokens; I processed in 256-wide chunks; each wave owns
//   16 intermediate columns (phase 1) and a 64-wide slice of D (phase 2).
// ---------------------------------------------------------------------------
#define BM   32
#define BI   256
#define XLD  1032       // 1024 + 8 halfs pad -> 16B row-offset, conflict-free b128
#define HLD  264        // 256 + 8 halfs pad
#define SMEM_BYTES (BM * XLD * 2 + BM * HLD * 2 + BM * 4 + BM * 4)

template <typename WT>
__global__ __launch_bounds__(512)
void moe_ffn_kernel(const float* __restrict__ x,
                    const WT*    __restrict__ w1,
                    const WT*    __restrict__ w3,
                    const WT*    __restrict__ w2,
                    const int*   __restrict__ cnt,
                    const int*   __restrict__ tokList,
                    const float* __restrict__ wList,
                    float*       __restrict__ out) {
    extern __shared__ char smem[];
    _Float16* Xs  = (_Float16*)smem;                          // [BM][XLD]
    _Float16* Hs  = (_Float16*)(smem + BM * XLD * 2);         // [BM][HLD]
    int*      toks = (int*)(smem + BM * XLD * 2 + BM * HLD * 2);
    float*    wts  = (float*)(toks + BM);

    const int e    = blockIdx.y;
    const int cnte = cnt[e];
    const int row0 = blockIdx.x * BM;
    if (row0 >= cnte) return;                 // uniform: whole block exits

    const int tid  = threadIdx.x;
    const int wave = tid >> 5;                // 0..15
    const int lane = tid & 31;
    const int n    = lane & 15;
    const int hi   = lane >> 4;

    if (tid < BM) {
        int r = row0 + tid;
        if (r < cnte) { toks[tid] = tokList[e * TOK + r]; wts[tid] = wList[e * TOK + r]; }
        else          { toks[tid] = tokList[e * TOK + row0]; wts[tid] = 0.f; }
    }
    __syncthreads();

    // stage X tile as f16: 16 threads per row, 16 float4 each (32 rows x 1024)
    {
        const int row = tid >> 4, tpr = tid & 15;
        const float* xr = x + (size_t)toks[row] * DDIM;
#pragma unroll
        for (int c = 0; c < 16; ++c) {
            const int col = (tpr + c * 16) * 4;
            const v4f f = *(const v4f*)(xr + col);
            v4h h; h[0] = (_Float16)f[0]; h[1] = (_Float16)f[1];
                   h[2] = (_Float16)f[2]; h[3] = (_Float16)f[3];
            *(v4h*)(&Xs[row * XLD + col]) = h;
        }
    }
    __syncthreads();

    v8f y[2][4];
#pragma unroll
    for (int mh = 0; mh < 2; ++mh)
#pragma unroll
        for (int nt = 0; nt < 4; ++nt) y[mh][nt] = (v8f){};

    for (int ic = 0; ic < IDIM / BI; ++ic) {          // 16 chunks
        // -------- phase 1: H chunk, this wave's 16 i-columns ---------------
        const int iBase = ic * BI + wave * 16;
        const WT* w1row = w1 + ((size_t)e * IDIM + iBase + n) * DDIM + hi * 16;
        const WT* w3row = w3 + ((size_t)e * IDIM + iBase + n) * DDIM + hi * 16;

        v8f a1[2] = {(v8f){}, (v8f){}};
        v8f a3[2] = {(v8f){}, (v8f){}};
#pragma unroll 4
        for (int ks = 0; ks < DDIM / 32; ++ks) {
            const int k0 = ks * 32;
            const v16h aLo = lds_afrag(Xs + n * XLD,        k0, hi);   // tokens 0..15
            const v16h aHi = lds_afrag(Xs + (16 + n) * XLD, k0, hi);   // tokens 16..31
            const v16h b1  = load_bfrag(w1row + k0);
            const v16h b3  = load_bfrag(w3row + k0);
            a1[0] = __builtin_amdgcn_wmma_f32_16x16x32_f16(false, aLo, false, b1,
                                                           (short)0, a1[0], false, false);
            a1[1] = __builtin_amdgcn_wmma_f32_16x16x32_f16(false, aHi, false, b1,
                                                           (short)0, a1[1], false, false);
            a3[0] = __builtin_amdgcn_wmma_f32_16x16x32_f16(false, aLo, false, b3,
                                                           (short)0, a3[0], false, false);
            a3[1] = __builtin_amdgcn_wmma_f32_16x16x32_f16(false, aHi, false, b3,
                                                           (short)0, a3[1], false, false);
        }

        __syncthreads();                      // prior phase-2 done reading Hs
#pragma unroll
        for (int mh = 0; mh < 2; ++mh) {
#pragma unroll
            for (int j = 0; j < 8; ++j) {
                const float z = a1[mh][j];
                const float h = z * (1.0f / (1.0f + __expf(-z))) * a3[mh][j];
                Hs[(mh * 16 + j + 8 * hi) * HLD + wave * 16 + n] = (_Float16)h;
            }
        }
        __syncthreads();                      // full Hs visible to all waves

        // -------- phase 2: Y += H @ w2-slice (64 D columns per wave) --------
#pragma unroll
        for (int ks = 0; ks < BI / 32; ++ks) {
            const int k0 = ks * 32;
            const v16h aLo = lds_afrag(Hs + n * HLD,        k0, hi);
            const v16h aHi = lds_afrag(Hs + (16 + n) * HLD, k0, hi);
            const int iidx = ic * BI + k0 + hi * 16;
#pragma unroll
            for (int nt = 0; nt < 4; ++nt) {
                const int dcol = wave * 64 + nt * 16 + n;
                const v16h b2 = load_bfrag(w2 + ((size_t)e * DDIM + dcol) * IDIM + iidx);
                y[0][nt] = __builtin_amdgcn_wmma_f32_16x16x32_f16(false, aLo, false, b2,
                                                                  (short)0, y[0][nt], false, false);
                y[1][nt] = __builtin_amdgcn_wmma_f32_16x16x32_f16(false, aHi, false, b2,
                                                                  (short)0, y[1][nt], false, false);
            }
        }
    }

    // epilogue: out[t, d] += routing_weight[t] * y
#pragma unroll
    for (int mh = 0; mh < 2; ++mh) {
#pragma unroll
        for (int nt = 0; nt < 4; ++nt) {
            const int dcol = wave * 64 + nt * 16 + n;
#pragma unroll
            for (int j = 0; j < 8; ++j) {
                const int   m  = mh * 16 + j + 8 * hi;
                const float wt = wts[m];
                if (wt != 0.f) {
                    atomicAdd(&out[(size_t)toks[m] * DDIM + dcol], y[mh][nt][j] * wt);
                }
            }
        }
    }
}

// ---------------------------------------------------------------------------
// Host launcher
// ---------------------------------------------------------------------------
#define WELEM  ((size_t)NEXP * IDIM * DDIM)      // elements per weight matrix
#define LISTS_BYTES (256 + (size_t)NEXP * TOK * 4 * 2)
#define WCVT_OFF    (1u << 20)                   // f16 weights start at 1 MB
#define WS_F16_NEEDED (WCVT_OFF + 3 * WELEM * sizeof(_Float16))

extern "C" void kernel_launch(void* const* d_in, const int* in_sizes, int n_in,
                              void* d_out, int out_size, void* d_ws, size_t ws_size,
                              hipStream_t stream) {
    const float* x  = (const float*)d_in[0];   // [2,2048,1024]
    const float* gw = (const float*)d_in[1];   // [8,1024]
    const float* w1 = (const float*)d_in[2];   // [8,4096,1024]
    const float* w3 = (const float*)d_in[3];   // [8,4096,1024]
    const float* w2 = (const float*)d_in[4];   // [8,1024,4096]

    float* out        = (float*)d_out;                       // [TOK, DDIM]
    float* logits_out = (float*)d_out + (size_t)TOK * DDIM;  // [TOK, NEXP]

    int*   cnt     = (int*)d_ws;                                   // 8 ints
    int*   tokList = (int*)((char*)d_ws + 256);                    // NEXP*TOK ints
    float* wList   = (float*)((char*)d_ws + 256 + NEXP * TOK * 4); // NEXP*TOK floats

    hipMemsetAsync(cnt, 0, NEXP * sizeof(int), stream);
    hipMemsetAsync(out, 0, (size_t)TOK * DDIM * sizeof(float), stream);

    router_kernel<<<TOK / 8, 256, 0, stream>>>(x, gw, logits_out, cnt, tokList, wList);

    dim3 grid(TOK / BM, NEXP);

    if (ws_size >= WS_F16_NEEDED) {
        // pre-convert weights to f16 once (memory-bound, ~600 MB of traffic)
        _Float16* w1h = (_Float16*)((char*)d_ws + WCVT_OFF);
        _Float16* w3h = w1h + WELEM;
        _Float16* w2h = w3h + WELEM;
        const int n4 = (int)(WELEM / 4);
        const int cblocks = (n4 + 255) / 256;
        cvt_f32_to_f16<<<cblocks, 256, 0, stream>>>(w1, w1h, n4);
        cvt_f32_to_f16<<<cblocks, 256, 0, stream>>>(w3, w3h, n4);
        cvt_f32_to_f16<<<cblocks, 256, 0, stream>>>(w2, w2h, n4);
        moe_ffn_kernel<_Float16><<<grid, 512, SMEM_BYTES, stream>>>(
            x, w1h, w3h, w2h, cnt, tokList, wList, out);
    } else {
        // fallback: convert inline in the hot loop
        moe_ffn_kernel<float><<<grid, 512, SMEM_BYTES, stream>>>(
            x, w1, w3, w2, cnt, tokList, wList, out);
    }
}
